// ealstm_82549271429808
// MI455X (gfx1250) — matrix operand order repaired and checked by reading, not compiled, not run on hardware
//
#include <hip/hip_runtime.h>
#include <hip/hip_bf16.h>

typedef _Float16 v16h __attribute__((ext_vector_type(16)));
typedef _Float16 v8h  __attribute__((ext_vector_type(8)));
typedef float    v8f  __attribute__((ext_vector_type(8)));

#define NT_STEPS 365
#define DD 32
#define HH 256
#define KK 288          // augmented K = H + Dd
#define GG 768          // 3*H
#define NTILES 48       // GG/16
#define KBLK 9          // KK/32  (kb 0..7 = W_hh, kb 8 = W_ih)

// ---------------------------------------------------------------------------
// Pack [W_hh(256x768); W_ih(32x768)] fp32 -> f16 B-operand fragments.
// B fragment (16x16x32 wmma): lane l, element e -> K = (e&7) + (l>>4)*8 + (e>>3)*16, N = l&15
// dst index: (((nt*9 + kb)*32 + l)*16 + e)
// ---------------------------------------------------------------------------
__global__ void pack_w_kernel(const float* __restrict__ whh,
                              const float* __restrict__ wih,
                              _Float16* __restrict__ dst) {
    int idx = blockIdx.x * blockDim.x + threadIdx.x;   // 48*9*32*16 = 221184
    if (idx >= NTILES * KBLK * 32 * 16) return;
    int e   = idx & 15;
    int l   = (idx >> 4) & 31;
    int rem = idx >> 9;
    int kb  = rem % KBLK;
    int nt  = rem / KBLK;
    int Ki  = (e & 7) + ((l >> 4) * 8) + ((e >> 3) * 16); // 0..31 within block
    int n   = nt * 16 + (l & 15);
    float v = (kb < 8) ? whh[(kb * 32 + Ki) * GG + n]
                       : wih[Ki * GG + n];
    dst[idx] = (_Float16)v;
}

// ---------------------------------------------------------------------------
// Persistent fused EA-LSTM. One block = 16 batch rows, 8 waves, loops over T.
// All 54 combined-weight fragments live in registers for all 365 steps
// (432 VGPRs); per-element recurrent state (c, i_gate, W_out, bias) in LDS.
// ---------------------------------------------------------------------------
__global__ __launch_bounds__(256) void ealstm_main_kernel(
    const float* __restrict__ xdyn,     // [B, T, 32]
    const float* __restrict__ xstat,    // [B, 27]
    const _Float16* __restrict__ wP,    // packed [W_hh; W_ih] fragments
    const float* __restrict__ bias,     // [768]
    const float* __restrict__ bias_s,   // [256]
    const float* __restrict__ wsh,      // [27, 256]
    const float* __restrict__ wout_g,   // [256, 1]
    const float* __restrict__ bout_g,   // [1]
    float* __restrict__ out)            // [B, T, 1]
{
    __shared__ __align__(32) _Float16 gatesh[16 * GG];  // 24 KB (staged gates, f16)
    __shared__ __align__(32) float    cls   [16 * HH];  // 16 KB (cell state, f32)
    __shared__ __align__(32) _Float16 hxbuf [16 * KK];  //  9 KB ([h | x_t], f16, A operand)
    __shared__ __align__(32) _Float16 igl   [16 * HH];  //  8 KB (input gate, f16)
    __shared__ float bsl[GG];                           //  3 KB (gate bias)
    __shared__ float wol[HH];                           //  1 KB (W_out)
    __shared__ float outpart[16];

    const int tid  = threadIdx.x;
    const int lane = tid & 31;
    const int w    = tid >> 5;          // wave id, 0..7
    const int b0   = blockIdx.x * 16;

    // elementwise ownership: thread -> (row em, 16 cols starting at jb)
    const int em = tid >> 4;            // 0..15
    const int jb = (tid & 15) * 16;     // 0..240

    // ---- one-time: i_gate = sigmoid(x_static @ W_sh + bias_s) -> LDS (f16)
    {
        float xs[27];
        #pragma unroll
        for (int d = 0; d < 27; d++) xs[d] = xstat[(b0 + em) * 27 + d];
        #pragma unroll
        for (int jj = 0; jj < 16; jj++) {
            float s = bias_s[jb + jj];
            #pragma unroll
            for (int d = 0; d < 27; d++) s += xs[d] * wsh[d * HH + jb + jj];
            igl[em * HH + jb + jj] = (_Float16)(1.0f / (1.0f + __expf(-s)));
            cls[em * HH + jb + jj] = 0.0f;
        }
    }
    wol[tid] = wout_g[tid & 255];
    for (int e = tid; e < GG; e += 256) bsl[e] = bias[e];
    for (int e = tid; e < 16 * KK; e += 256) hxbuf[e] = (_Float16)0.0f;
    if (tid < 16) outpart[tid] = 0.0f;
    const float bout = bout_g[0];

    // ---- register-resident weights: 6 tiles x 9 K-blocks = 54 fragments
    v16h wF[6 * KBLK];                  // 432 VGPRs
    #pragma unroll
    for (int i = 0; i < 6; i++) {
        int nt = w * 6 + i;
        #pragma unroll
        for (int kb = 0; kb < KBLK; kb++) {
            wF[i * KBLK + kb] =
                *(const v16h*)(wP + (size_t)((nt * KBLK + kb) * 32 + lane) * 16);
        }
    }
    __syncthreads();

    const int m16  = lane & 15;
    const int hi   = lane >> 4;
    const int koff = hi * 8;
    const int n16  = lane & 15;

    for (int t = 0; t < NT_STEPS; t++) {
        // keep loop-invariant LDS reads (igl/wol/bsl) from being hoisted to regs
        asm volatile("" ::: "memory");

        // ---- stage x_t tile into augmented A buffer columns 256..287
        #pragma unroll
        for (int e = tid; e < 16 * DD; e += 256) {
            int m = e >> 5, k = e & 31;
            hxbuf[m * KK + HH + k] =
                (_Float16)xdyn[((size_t)(b0 + m) * NT_STEPS + t) * DD + k];
        }
        __syncthreads();

        // ---- three passes of 2 N-tiles (caps live accumulators at 16 VGPRs)
        #pragma unroll
        for (int p = 0; p < 3; p++) {
            v8f acc[2];
            #pragma unroll
            for (int i = 0; i < 2; i++) {
                #pragma unroll
                for (int r = 0; r < 8; r++) acc[i][r] = 0.0f;
            }
            #pragma unroll
            for (int kb = 0; kb < KBLK; kb++) {
                v16h a;
                {
                    v8h lo = *(const v8h*)(&hxbuf[m16 * KK + kb * 32 + koff]);
                    v8h hu = *(const v8h*)(&hxbuf[m16 * KK + kb * 32 + 16 + koff]);
                    #pragma unroll
                    for (int q = 0; q < 8; q++) { a[q] = lo[q]; a[8 + q] = hu[q]; }
                }
                #pragma unroll
                for (int i = 0; i < 2; i++) {
                    acc[i] = __builtin_amdgcn_wmma_f32_16x16x32_f16(
                        false, a, false, wF[(p * 2 + i) * KBLK + kb],
                        (short)0, acc[i], false, false);
                }
            }
            // stage gate tiles (D layout: lane col = l&15, rows r + hi*8)
            #pragma unroll
            for (int i = 0; i < 2; i++) {
                int col  = (w * 6 + p * 2 + i) * 16 + n16;
                float bb = bsl[col];
                #pragma unroll
                for (int r = 0; r < 8; r++) {
                    gatesh[(r + hi * 8) * GG + col] = (_Float16)(acc[i][r] + bb);
                }
            }
            __builtin_amdgcn_sched_barrier(0);   // keep pass lifetimes disjoint
        }
        __syncthreads();

        // ---- elementwise LSTM cell + fused output projection
        float hpart = 0.0f;
        #pragma unroll
        for (int jj = 0; jj < 16; jj++) {
            int j   = jb + jj;
            float f = (float)gatesh[em * GG + j];
            float o = (float)gatesh[em * GG + HH + j];
            float g = (float)gatesh[em * GG + 2 * HH + j];
            float sf = 1.0f / (1.0f + __expf(-f));
            float so = 1.0f / (1.0f + __expf(-o));
            float c  = sf * cls[em * HH + j] + (float)igl[em * HH + j] * tanhf(g);
            cls[em * HH + j] = c;
            float h  = so * tanhf(c);
            hxbuf[em * KK + j] = (_Float16)h;
            hpart += h * wol[j];
        }
        atomicAdd(&outpart[em], hpart);
        __syncthreads();

        if (tid < 16) {
            out[(size_t)(b0 + tid) * NT_STEPS + t] = outpart[tid] + bout;
            outpart[tid] = 0.0f;
        }
        // reset is separated from next step's atomics by two barriers
    }
}

extern "C" void kernel_launch(void* const* d_in, const int* in_sizes, int n_in,
                              void* d_out, int out_size, void* d_ws, size_t ws_size,
                              hipStream_t stream) {
    const float* xdyn   = (const float*)d_in[0];
    const float* xstat  = (const float*)d_in[1];
    const float* wih    = (const float*)d_in[2];
    const float* whh    = (const float*)d_in[3];
    const float* wsh    = (const float*)d_in[4];
    const float* bias   = (const float*)d_in[5];
    const float* bias_s = (const float*)d_in[6];
    const float* wout   = (const float*)d_in[7];
    const float* bout   = (const float*)d_in[8];
    float* out = (float*)d_out;

    _Float16* wP = (_Float16*)d_ws;     // 432 KB packed combined weights

    const int packN = NTILES * KBLK * 32 * 16;          // 221184
    pack_w_kernel<<<(packN + 255) / 256, 256, 0, stream>>>(whh, wih, wP);

    ealstm_main_kernel<<<64, 256, 0, stream>>>(
        xdyn, xstat, wP, bias, bias_s, wsh, wout, bout, out);
}
